// get_model_50002009260617
// MI455X (gfx1250) — matrix operand order repaired and checked by reading, not compiled, not run on hardware
//
#include <hip/hip_runtime.h>
#include <hip/hip_bf16.h>
#include <stdint.h>

// ---------------- problem constants (from reference) ----------------
static constexpr int Bb = 2, Nn = 16384, KP = 2048, NC = 14;
#define BN_INV 0.9999950000374996f // 1/sqrt(1+1e-5)

typedef __bf16 bf16;
typedef bf16  v16bf __attribute__((ext_vector_type(16)));
typedef float v8f   __attribute__((ext_vector_type(8)));
typedef float v4f   __attribute__((ext_vector_type(4)));

#define GTID (int)(blockIdx.x * blockDim.x + threadIdx.x)
static inline int cdiv(int a, int b) { return (a + b - 1) / b; }
static inline int rup32(int x) { return (x + 31) & ~31; }

// =====================================================================
// weight transpose + f32->bf16 pack + zero-pad: Wt[Ncp][Kp] (bf16)
// Wt[n][k] = (bf16)W[k][n]
// =====================================================================
__global__ void transw_kernel(const float* __restrict__ W, bf16* __restrict__ Wt,
                              int K, int Nc, int Kp, int Ncp, int total)
{
    int gid = GTID; if (gid >= total) return;     // total = Ncp*Kp
    int k = gid % Kp;
    int n = gid / Kp;
    float v = (n < Nc && k < K) ? W[(size_t)k * Nc + n] : 0.0f;
    Wt[gid] = (bf16)v;
}

// =====================================================================
// WMMA GEMM: Y[M,Ncp] = act((X[M,Kp] @ W + bias)*scale + beta)
// X: f32, padded (stride Kp, zeros in pad). Wt: bf16, transposed+padded.
// One wave per 16x32 output tile, branch-free k-loop, unrolled x2:
//   A: 4x b128 f32 loads + cvt_pk_bf16; B: direct v16bf (32B) loads.
//   4x v_wmma_f32_16x16x32_bf16 per unrolled body.
// M must be a multiple of 16 (guaranteed by caller).
// =====================================================================
__global__ void gemm_wmma(const float* __restrict__ X, const bf16* __restrict__ Wt,
                          const float* __restrict__ bias, const float* __restrict__ gamma,
                          const float* __restrict__ beta, float* __restrict__ Y,
                          int Kp, int Nc, int Ncp, int relu)
{
    const int lane   = threadIdx.x & 31;
    const int laneLo = lane & 15;
    const int laneHi = lane >> 4;
    const int tm     = blockIdx.x << 4;
    const int tn     = blockIdx.y << 5;

    const float* xrow = X  + (size_t)(tm + laneLo) * Kp;
    const bf16* b0row = Wt + (size_t)(tn + laneLo) * Kp;
    const bf16* b1row = Wt + (size_t)(tn + 16 + laneLo) * Kp;

    v8f acc0 = {}, acc1 = {};
#pragma unroll 2
    for (int k0 = 0; k0 < Kp; k0 += 32) {
        __builtin_prefetch(xrow + k0 + 128, 0, 1); // global_prefetch_b8 (speculative)

        // A fragment: row = tm+laneLo; runs [k0+laneHi*16, +8) and [k0+laneHi*16+8, +8)
        const v4f* ap0 = (const v4f*)(xrow + k0 + (laneHi << 4));
        const v4f* ap1 = (const v4f*)(xrow + k0 + 8 + (laneHi << 4));
        v4f a0 = ap0[0], a1 = ap0[1], a2 = ap1[0], a3 = ap1[1];

        // B fragments (bf16, contiguous 16 elems = 32B each)
        v16bf b0 = *(const v16bf*)(b0row + k0 + (laneHi << 4));
        v16bf b1 = *(const v16bf*)(b1row + k0 + (laneHi << 4));

        v16bf a;
#pragma unroll
        for (int i = 0; i < 4; ++i) {
            a[i]      = (bf16)a0[i];
            a[4 + i]  = (bf16)a1[i];
            a[8 + i]  = (bf16)a2[i];
            a[12 + i] = (bf16)a3[i];
        }
        acc0 = __builtin_amdgcn_wmma_f32_16x16x32_bf16(false, a, false, b0, (short)0, acc0, false, false);
        acc1 = __builtin_amdgcn_wmma_f32_16x16x32_bf16(false, a, false, b1, (short)0, acc1, false, false);
    }

    // epilogue: C/D layout -> lane&15 = col, laneHi*8+e = row
    const int col0 = tn + laneLo;
    const int col1 = col0 + 16;
    float bb0 = 0.f, sc0 = 1.f, bt0 = 0.f, bb1 = 0.f, sc1 = 1.f, bt1 = 0.f;
    bool v0ok = col0 < Nc, v1ok = col1 < Nc;
    if (v0ok) {
        bb0 = bias ? bias[col0] : 0.f;
        sc0 = gamma ? (BN_INV * gamma[col0]) : 1.f;
        bt0 = beta ? beta[col0] : 0.f;
    }
    if (v1ok) {
        bb1 = bias ? bias[col1] : 0.f;
        sc1 = gamma ? (BN_INV * gamma[col1]) : 1.f;
        bt1 = beta ? beta[col1] : 0.f;
    }
#pragma unroll
    for (int e = 0; e < 8; ++e) {
        int m = tm + (laneHi << 3) + e;
        float v0 = (acc0[e] + bb0) * sc0 + bt0;
        float v1 = (acc1[e] + bb1) * sc1 + bt1;
        if (relu) { v0 = fmaxf(v0, 0.f); v1 = fmaxf(v1, 0.f); }
        if (!v0ok) v0 = 0.f;
        if (!v1ok) v1 = 0.f;
        Y[(size_t)m * Ncp + col0] = v0;
        Y[(size_t)m * Ncp + col1] = v1;
    }
}

// =====================================================================
// (B,9,N) -> (B,N,9)
// =====================================================================
__global__ void transpose_kernel(const float* __restrict__ in, float* __restrict__ out, int total)
{
    int gid = GTID; if (gid >= total) return;
    int c = gid % 9; int bn = gid / 9; int n = bn % Nn; int b = bn / Nn;
    out[gid] = in[((size_t)b * 9 + c) * Nn + n];
}

// =====================================================================
// farthest point sampling: one block per batch. xyz strided by Cx.
// =====================================================================
__global__ void fps_kernel(const float* __restrict__ xyz, int Nl, int Cx, int npoint,
                           int* __restrict__ idx_out, float* __restrict__ dist)
{
    int b = blockIdx.x;
    const float* pb = xyz + (size_t)b * Nl * Cx;
    float* db = dist + (size_t)b * Nl;
    __shared__ float smax[256];
    __shared__ int   simax[256];
    __shared__ int   s_far;
    int t = threadIdx.x;
    for (int i = t; i < Nl; i += 256) db[i] = 1e10f;
    if (t == 0) s_far = 0;
    __syncthreads();
    for (int it = 0; it < npoint; ++it) {
        int far = s_far;
        if (t == 0) idx_out[b * npoint + it] = far;
        float cx = pb[(size_t)far * Cx + 0];
        float cy = pb[(size_t)far * Cx + 1];
        float cz = pb[(size_t)far * Cx + 2];
        float best = -1.0f; int bi = 0;
        for (int i = t; i < Nl; i += 256) {
            float dx = pb[(size_t)i * Cx + 0] - cx;
            float dy = pb[(size_t)i * Cx + 1] - cy;
            float dz = pb[(size_t)i * Cx + 2] - cz;
            float d  = dx * dx + dy * dy + dz * dz;
            float nd = fminf(db[i], d);
            db[i] = nd;
            if (nd > best) { best = nd; bi = i; }
        }
        smax[t] = best; simax[t] = bi;
        __syncthreads();
        for (int s = 128; s > 0; s >>= 1) {
            if (t < s && smax[t + s] > smax[t]) { smax[t] = smax[t + s]; simax[t] = simax[t + s]; }
            __syncthreads();
        }
        if (t == 0) s_far = simax[0];
        __syncthreads();
    }
}

// gather xyz (3ch) by index from strided source
__global__ void gather3_kernel(const float* __restrict__ src, int Cx, int Nl,
                               const int* __restrict__ idx, int npoint,
                               float* __restrict__ out, int total)
{
    int gid = GTID; if (gid >= total) return;     // total = Bb*npoint
    int b = gid / npoint;
    int j = idx[gid];
    const float* p = src + ((size_t)b * Nl + j) * Cx;
    out[gid * 3 + 0] = p[0];
    out[gid * 3 + 1] = p[1];
    out[gid * 3 + 2] = p[2];
}

// =====================================================================
// ball query (first-K indices ascending, padded with first)
// =====================================================================
__global__ void query_ball_kernel(const float* __restrict__ xyz, int Cx, int Nl,
                                  const float* __restrict__ new_xyz, int npoint,
                                  float r2, int K, int* __restrict__ gidx, int total)
{
    int gid = GTID; if (gid >= total) return;     // total = Bb*npoint
    int b = gid / npoint;
    const float* c  = new_xyz + (size_t)gid * 3;
    const float* pb = xyz + (size_t)b * Nl * Cx;
    int* g = gidx + (size_t)gid * K;
    int cnt = 0, first = -1;
    for (int j = 0; j < Nl; ++j) {
        float dx = pb[(size_t)j * Cx + 0] - c[0];
        float dy = pb[(size_t)j * Cx + 1] - c[1];
        float dz = pb[(size_t)j * Cx + 2] - c[2];
        float d  = dx * dx + dy * dy + dz * dz;
        if (d <= r2) {
            if (first < 0) first = j;
            g[cnt++] = j;
            if (cnt == K) break;
        }
    }
    if (first < 0) first = Nl - 1;
    for (; cnt < K; ++cnt) g[cnt] = first;
}

// h[row] = concat(pts[idx] (C), xyz[idx]-new_xyz (3)), zero-pad to Kp
__global__ void group_feats_kernel(const float* __restrict__ pts, int C,
                                   const float* __restrict__ xyz, int Cx,
                                   const float* __restrict__ new_xyz,
                                   const int* __restrict__ gidx, int Nl,
                                   int npoint, int K, float* __restrict__ h, int Kp, int total)
{
    int gid = GTID; if (gid >= total) return;     // total = Bb*npoint*K
    int bm = gid / K;
    int b  = bm / npoint;
    int j  = gidx[gid];
    float* r = h + (size_t)gid * Kp;
    const float* f = pts + ((size_t)b * Nl + j) * C;
    for (int c = 0; c < C; ++c) r[c] = f[c];
    const float* xp = xyz + ((size_t)b * Nl + j) * Cx;
    const float* nx = new_xyz + (size_t)bm * 3;
    r[C + 0] = xp[0] - nx[0];
    r[C + 1] = xp[1] - nx[1];
    r[C + 2] = xp[2] - nx[2];
    for (int c = C + 3; c < Kp; ++c) r[c] = 0.0f;
}

// out[bm, coff+c] = max_k h[(bm*K+k)*Cout + c]   (Cout already mult of 32)
__global__ void maxpool_kernel(const float* __restrict__ h, int K, int Cout,
                               float* __restrict__ out, int Ctot, int coff, int total)
{
    int gid = GTID; if (gid >= total) return;     // total = Bb*npoint*Cout
    int c  = gid % Cout;
    int bm = gid / Cout;
    const float* p = h + (size_t)bm * K * Cout + c;
    float m = p[0];
    for (int k = 1; k < K; ++k) m = fmaxf(m, p[(size_t)k * Cout]);
    out[(size_t)bm * Ctot + coff + c] = m;
}

// =====================================================================
// feature propagation: 3-NN inverse-distance interp + concat, pad to Kp
// =====================================================================
__global__ void fp_concat_kernel(const float* __restrict__ xyz1, int C1x,
                                 const float* __restrict__ xyz2,
                                 const float* __restrict__ pts1, int C1,
                                 const float* __restrict__ pts2, int C2, int C2s,
                                 int N1, int N2, float* __restrict__ h, int Kp, int total)
{
    int gid = GTID; if (gid >= total) return;     // total = Bb*N1
    int b = gid / N1, i = gid % N1;
    const float* p1 = xyz1 + ((size_t)b * N1 + i) * C1x;
    float x = p1[0], y = p1[1], z = p1[2];
    float d0 = 1e30f, d1 = 1e30f, d2 = 1e30f;
    int   i0 = 0, i1 = 0, i2 = 0;
    const float* x2 = xyz2 + (size_t)b * N2 * 3;
    for (int j = 0; j < N2; ++j) {
        float dx = x2[j * 3 + 0] - x;
        float dy = x2[j * 3 + 1] - y;
        float dz = x2[j * 3 + 2] - z;
        float d  = dx * dx + dy * dy + dz * dz;
        if (d < d0)      { d2 = d1; i2 = i1; d1 = d0; i1 = i0; d0 = d; i0 = j; }
        else if (d < d1) { d2 = d1; i2 = i1; d1 = d;  i1 = j; }
        else if (d < d2) { d2 = d;  i2 = j; }
    }
    float w0 = 1.0f / (d0 + 1e-8f), w1 = 1.0f / (d1 + 1e-8f), w2 = 1.0f / (d2 + 1e-8f);
    float ws = w0 + w1 + w2; w0 /= ws; w1 /= ws; w2 /= ws;
    float* hr = h + (size_t)gid * Kp;
    const float* f1 = pts1 + ((size_t)b * N1 + i) * C1;
    for (int c = 0; c < C1; ++c) hr[c] = f1[c];
    const float* f2 = pts2 + (size_t)b * N2 * C2s;
    for (int c = 0; c < C2; ++c)
        hr[C1 + c] = f2[(size_t)i0 * C2s + c] * w0 +
                     f2[(size_t)i1 * C2s + c] * w1 +
                     f2[(size_t)i2 * C2s + c] * w2;
    for (int c = C1 + C2; c < Kp; ++c) hr[c] = 0.0f;
}

// =====================================================================
// final head: packed (dist,idx) keys; exact sorted top-KP
// =====================================================================
__global__ void dist_keys_kernel(const float* __restrict__ cent,
                                 const float* __restrict__ xyz1, int C1x,
                                 unsigned long long* __restrict__ keys, int total)
{
    int gid = GTID; if (gid >= total) return;     // total = Bb*NC*Nn
    int j  = gid % Nn;
    int bc = gid / Nn;
    int b  = bc / NC;
    const float* c = cent + (size_t)bc * 3;
    const float* p = xyz1 + ((size_t)b * Nn + j) * C1x;
    float dx = p[0] - c[0], dy = p[1] - c[1], dz = p[2] - c[2];
    float d  = dx * dx + dy * dy + dz * dz;
    keys[gid] = ((unsigned long long)__float_as_uint(d) << 32) | (unsigned)j;
}

__global__ void topk_kernel(const unsigned long long* __restrict__ keys,
                            int Nl, int Kp_, int* __restrict__ pidx, float* __restrict__ pdist)
{
    int bc = blockIdx.x;
    const unsigned long long* kb = keys + (size_t)bc * Nl;
    __shared__ unsigned long long sk[256];
    __shared__ unsigned long long s_prev;
    int t = threadIdx.x;
    if (t == 0) s_prev = 0ULL;
    __syncthreads();
    for (int it = 0; it < Kp_; ++it) {
        unsigned long long prev = s_prev;
        unsigned long long best = ~0ULL;
        for (int j = t; j < Nl; j += 256) {
            unsigned long long k = kb[j];
            if (k >= prev && k < best) best = k;
        }
        sk[t] = best;
        __syncthreads();
        for (int s = 128; s > 0; s >>= 1) {
            if (t < s && sk[t + s] < sk[t]) sk[t] = sk[t + s];
            __syncthreads();
        }
        if (t == 0) {
            unsigned long long sel = sk[0];
            pidx[(size_t)bc * Kp_ + it]  = (int)(sel & 0xffffffffu);
            pdist[(size_t)bc * Kp_ + it] = __uint_as_float((unsigned)(sel >> 32));
            s_prev = sel + 1ULL;
        }
        __syncthreads();
    }
}

// prop row = [gxyz(3), feats(32), dist(1)], padded to 64
__global__ void final_prop_kernel(const float* __restrict__ cent,
                                  const float* __restrict__ xyz1, int C1x,
                                  const float* __restrict__ feats,
                                  const int* __restrict__ pidx,
                                  const float* __restrict__ pdist,
                                  float* __restrict__ prop, int total)
{
    int gid = GTID; if (gid >= total) return;     // total = Bb*NC*KP
    int bc = gid / KP;
    int b  = bc / NC;
    int j  = pidx[gid];
    const float* c  = cent + (size_t)bc * 3;
    const float* xp = xyz1 + ((size_t)b * Nn + j) * C1x;
    float* r = prop + (size_t)gid * 64;
    r[0] = xp[0] - c[0];
    r[1] = xp[1] - c[1];
    r[2] = xp[2] - c[2];
    const float* f = feats + ((size_t)b * Nn + j) * 32;
    for (int cc = 0; cc < 32; ++cc) r[3 + cc] = f[cc];
    r[35] = pdist[gid];
    for (int cc = 36; cc < 64; ++cc) r[cc] = 0.0f;
}

__global__ void logsoftmax_kernel(const float* __restrict__ h, int hs,
                                  float* __restrict__ out, int total)
{
    int gid = GTID; if (gid >= total) return;     // total = rows
    const float* r = h + (size_t)gid * hs;
    float mx = r[0];
    for (int c = 1; c < 16; ++c) mx = fmaxf(mx, r[c]);
    float sum = 0.0f;
    for (int c = 0; c < 16; ++c) sum += __expf(r[c] - mx);
    float ls = mx + __logf(sum);
    float* o = out + (size_t)gid * 16;
    for (int c = 0; c < 16; ++c) o[c] = r[c] - ls;
}

__global__ void pidx_out_kernel(const int* __restrict__ pidx, float* __restrict__ out, int total)
{
    int gid = GTID; if (gid >= total) return;
    out[gid] = (float)pidx[gid];
}

// =====================================================================
// host orchestration
// =====================================================================
static inline const float* ip(void* const* d_in, int i) { return (const float*)d_in[i]; }

static void launch_gemm(hipStream_t s, const float* X, const float* W, const float* b,
                        const float* g, const float* bt, float* Y,
                        int M, int K, int Nc, int relu, bf16* wt)
{
    int Kp = rup32(K), Ncp = rup32(Nc);
    int tw = Ncp * Kp;
    transw_kernel<<<cdiv(tw, 256), 256, 0, s>>>(W, wt, K, Nc, Kp, Ncp, tw);
    dim3 grid(M / 16, Ncp / 32);
    gemm_wmma<<<grid, 32, 0, s>>>(X, wt, b, g, bt, Y, Kp, Nc, Ncp, relu);
}

// mlp chain: d_in[pbase + 4*l + {w,b,g,beta}], relu+bn every layer.
// h has stride rup32(cin); each output has stride rup32(co).
static void run_mlp(hipStream_t s, void* const* d_in, int pbase, const float* h,
                    int M, int cin, const int* chans, int nl,
                    float* bufA, float* bufB, float* final_dst, bf16* wt)
{
    const float* cur = h; int c = cin;
    for (int l = 0; l < nl; ++l) {
        int co = chans[l];
        float* dst = (l == nl - 1) ? final_dst : ((cur == bufA) ? bufB : bufA);
        launch_gemm(s, cur, ip(d_in, pbase + 4 * l + 0), ip(d_in, pbase + 4 * l + 1),
                    ip(d_in, pbase + 4 * l + 2), ip(d_in, pbase + 4 * l + 3),
                    dst, M, c, co, 1, wt);
        cur = dst; c = co;
    }
}

static void sa_msg_host(hipStream_t s, void* const* d_in,
                        const float* xyz, int Cx, const float* pts, int C,
                        int Nl, int npoint, const float* radii, const int* Ks,
                        int pbase, const int* chans,
                        float* out_xyz, float* out_pts,
                        int* fps_idx, float* fps_dist, int* gidx,
                        float* bufA, float* bufB, bf16* wt)
{
    fps_kernel<<<Bb, 256, 0, s>>>(xyz, Nl, Cx, npoint, fps_idx, fps_dist);
    int totc = Bb * npoint;
    gather3_kernel<<<cdiv(totc, 256), 256, 0, s>>>(xyz, Cx, Nl, fps_idx, npoint, out_xyz, totc);
    int Ctot = 2 * chans[2];
    int Kp = rup32(C + 3);
    for (int sc = 0; sc < 2; ++sc) {
        int K = Ks[sc];
        float r2 = radii[sc] * radii[sc];
        query_ball_kernel<<<cdiv(totc, 128), 128, 0, s>>>(xyz, Cx, Nl, out_xyz, npoint, r2, K, gidx, totc);
        int M = Bb * npoint * K;
        group_feats_kernel<<<cdiv(M, 128), 128, 0, s>>>(pts, C, xyz, Cx, out_xyz, gidx, Nl, npoint, K, bufA, Kp, M);
        run_mlp(s, d_in, pbase + sc * 12, bufA, M, C + 3, chans, 3, bufA, bufB, bufB, wt);
        int tp = Bb * npoint * chans[2];
        maxpool_kernel<<<cdiv(tp, 256), 256, 0, s>>>(bufB, K, chans[2], out_pts, Ctot, sc * chans[2], tp);
    }
}

extern "C" void kernel_launch(void* const* d_in, const int* in_sizes, int n_in,
                              void* d_out, int out_size, void* d_ws, size_t ws_size,
                              hipStream_t stream)
{
    (void)in_sizes; (void)n_in; (void)out_size; (void)ws_size;
    const float* xyz_in = ip(d_in, 0);
    const float* cent   = ip(d_in, 1);

    // ---- bump allocator over workspace ----
    char* ws = (char*)d_ws; size_t off = 0;
    auto alloc = [&](size_t bytes) -> void* {
        void* p = ws + off;
        off += (bytes + 255) & ~(size_t)255;
        return p;
    };
    float* l0_pts   = (float*)alloc(sizeof(float) * Bb * Nn * 9);
    float* fps_dist = (float*)alloc(sizeof(float) * Bb * Nn);
    int*   fps_idx  = (int*)  alloc(sizeof(int)   * Bb * 1024);
    int*   gidx     = (int*)  alloc(sizeof(int)   * Bb * 1024 * 32);
    float* l1_xyz   = (float*)alloc(sizeof(float) * Bb * 1024 * 3);
    float* l2_xyz   = (float*)alloc(sizeof(float) * Bb * 512 * 3);
    float* l3_xyz   = (float*)alloc(sizeof(float) * Bb * 256 * 3);
    float* l1_pts   = (float*)alloc(sizeof(float) * Bb * 1024 * 64);
    float* l2_pts   = (float*)alloc(sizeof(float) * Bb * 512 * 256);
    float* l3_pts   = (float*)alloc(sizeof(float) * Bb * 256 * 512);
    float* l2f      = (float*)alloc(sizeof(float) * Bb * 512 * 256);
    float* l1f      = (float*)alloc(sizeof(float) * Bb * 1024 * 128);
    float* l0f      = (float*)alloc(sizeof(float) * Bb * Nn * 32);
    unsigned long long* keys = (unsigned long long*)alloc(sizeof(unsigned long long) * Bb * NC * Nn);
    int*   pidx     = (int*)  alloc(sizeof(int)   * Bb * NC * KP);
    float* pdist    = (float*)alloc(sizeof(float) * Bb * NC * KP);
    bf16*  wt       = (bf16*) alloc(sizeof(bf16)  * 262144);   // transposed bf16 weights (max 256x768)
    float* bufA     = (float*)alloc(sizeof(float) * 6000000);
    float* bufB     = (float*)alloc(sizeof(float) * 6000000);

    // ---- l0: transpose (B,9,N) -> (B,N,9) ----
    {
        int tot = Bb * Nn * 9;
        transpose_kernel<<<cdiv(tot, 256), 256, 0, stream>>>(xyz_in, l0_pts, tot);
    }

    // ---- SA levels (MSG) ----
    static const float r1[2] = {2.5f, 5.0f},  r2a[2] = {5.0f, 10.0f}, r3[2] = {10.0f, 20.0f};
    static const int   Ks[2] = {16, 32};
    static const int   ch1[3] = {9, 32, 32}, ch2[3] = {64, 64, 128}, ch3[3] = {256, 196, 256};
    sa_msg_host(stream, d_in, l0_pts, 9, l0_pts, 9,   Nn,   1024, r1,  Ks,  2, ch1, l1_xyz, l1_pts, fps_idx, fps_dist, gidx, bufA, bufB, wt);
    sa_msg_host(stream, d_in, l1_xyz, 3, l1_pts, 64,  1024, 512,  r2a, Ks, 26, ch2, l2_xyz, l2_pts, fps_idx, fps_dist, gidx, bufA, bufB, wt);
    sa_msg_host(stream, d_in, l2_xyz, 3, l2_pts, 256, 512,  256,  r3,  Ks, 50, ch3, l3_xyz, l3_pts, fps_idx, fps_dist, gidx, bufA, bufB, wt);

    // ---- feature propagation ----
    static const int fp3c[2] = {256, 256}, fp2c[2] = {128, 128}, fp1c[2] = {64, 32};
    {   // fp3: l2 <- l3   cin = 256+512 = 768
        int tot = Bb * 512;
        fp_concat_kernel<<<cdiv(tot, 128), 128, 0, stream>>>(l2_xyz, 3, l3_xyz, l2_pts, 256, l3_pts, 512, 512, 512, 256, bufA, rup32(768), tot);
        run_mlp(stream, d_in, 74, bufA, tot, 768, fp3c, 2, bufA, bufB, l2f, wt);
    }
    {   // fp2: l1 <- l2   cin = 64+256 = 320
        int tot = Bb * 1024;
        fp_concat_kernel<<<cdiv(tot, 128), 128, 0, stream>>>(l1_xyz, 3, l2_xyz, l1_pts, 64, l2f, 256, 256, 1024, 512, bufA, rup32(320), tot);
        run_mlp(stream, d_in, 82, bufA, tot, 320, fp2c, 2, bufA, bufB, l1f, wt);
    }
    {   // fp1: l0 <- l1   cin = 9+128 = 137 (pad 160)
        int tot = Bb * Nn;
        fp_concat_kernel<<<cdiv(tot, 128), 128, 0, stream>>>(l0_pts, 9, l1_xyz, l0_pts, 9, l1f, 128, 128, Nn, 1024, bufA, rup32(137), tot);
        run_mlp(stream, d_in, 90, bufA, tot, 137, fp1c, 2, bufA, bufB, l0f, wt);
    }

    // ---- final head: top-KP nearest per centroid ----
    {
        int tot = Bb * NC * Nn;
        dist_keys_kernel<<<cdiv(tot, 256), 256, 0, stream>>>(cent, l0_pts, 9, keys, tot);
        topk_kernel<<<Bb * NC, 256, 0, stream>>>(keys, Nn, KP, pidx, pdist);
        int rows = Bb * NC * KP;
        final_prop_kernel<<<cdiv(rows, 128), 128, 0, stream>>>(cent, l0_pts, 9, l0f, pidx, pdist, bufA, rows);
        // conv1 (+bn1+relu): 36->36 (Kp=64, Ncp=64); conv2: 36->16 (Ncp=32)
        launch_gemm(stream, bufA, ip(d_in, 98),  ip(d_in, 99),  ip(d_in, 100), ip(d_in, 101), bufB, rows, 36, 36, 1, wt);
        launch_gemm(stream, bufB, ip(d_in, 102), ip(d_in, 103), nullptr,       nullptr,       bufA, rows, 36, 16, 0, wt);
        logsoftmax_kernel<<<cdiv(rows, 256), 256, 0, stream>>>(bufA, 32, (float*)d_out, rows);
        pidx_out_kernel<<<cdiv(rows, 256), 256, 0, stream>>>(pidx, (float*)d_out + (size_t)rows * 16, rows);
    }
}
